// PaiNN_17059610100462
// MI455X (gfx1250) — compile-verified, matrix operand order checked by reading, not run on hardware
//
#include <hip/hip_runtime.h>
#include <hip/hip_bf16.h>
#include <math.h>

#define NN 10000
#define EE 160000
#define FF 128
#define GG 200

typedef __attribute__((ext_vector_type(16))) __bf16 v16bf;
typedef __attribute__((ext_vector_type(8)))  __bf16 bf16x8;
typedef __attribute__((ext_vector_type(8)))  float  v8f;

#if defined(__has_builtin)
#if __has_builtin(__builtin_amdgcn_tensor_load_to_lds)
#define HAVE_TDM 1
#endif
#if __has_builtin(__builtin_amdgcn_s_wait_tensorcnt)
#define HAVE_TENSORCNT 1
#endif
#endif

// ---------------------------------------------------------------------------
// WMMA helpers (CDNA5 wave32, V_WMMA_F32_16X16X32_BF16)
// A fragment: lane L (L<16) holds row M=L, K = {kc+0..7, kc+16..23};
//             lane L>=16 holds row M=L-16, K = {kc+8..15, kc+24..31}.
// B fragment: same pattern with "row" = output column N (weights packed [N][K]).
// C/D: VGPR i holds (M = i + 8*(lane>=16), N = lane&15).
// ---------------------------------------------------------------------------
__device__ __forceinline__ v16bf load_frag(const __bf16* __restrict__ buf,
                                           int stride, int row0, int kc) {
  const int lane = threadIdx.x & 31;
  const int half = lane >> 4;
  const __bf16* p = buf + (size_t)(row0 + (lane & 15)) * stride + kc;
  bf16x8 lo = *(const bf16x8*)(p + (half ? 8 : 0));
  bf16x8 hi = *(const bf16x8*)(p + (half ? 24 : 16));
  v16bf f;
#pragma unroll
  for (int i = 0; i < 8; ++i) { f[i] = lo[i]; f[i + 8] = hi[i]; }
  return f;
}

__device__ __forceinline__ v8f wmma_bf16(v16bf a, v16bf b, v8f c) {
  return __builtin_amdgcn_wmma_f32_16x16x32_bf16(
      /*neg_a=*/false, a, /*neg_b=*/false, b,
      /*c_mod=*/(short)0, c, /*reuse_a=*/false, /*reuse_b=*/false);
}

__device__ __forceinline__ float silu_f(float x) {
  return x / (1.f + __expf(-x));
}

__device__ __forceinline__ void atomic_add_f32(float* p, float v) {
  __hip_atomic_fetch_add(p, v, __ATOMIC_RELAXED, __HIP_MEMORY_SCOPE_AGENT);
}

#ifdef HAVE_TDM
typedef unsigned int u32x4 __attribute__((ext_vector_type(4)));
typedef int          i32x4 __attribute__((ext_vector_type(4)));
typedef int          i32x8 __attribute__((ext_vector_type(8)));

// 1-D TDM copy: n_elem bf16 elements from global -> LDS.
// D# per CDNA5 ISA: group0 = {count, lds_addr,
// global_addr, type=2}; group1 = {data_size=1(2B), tensor_dim0=n, tensor_dim1=1,
// tile_dim0=n, tile_dim1=1, tensor_dim0_stride=n}.  LDS address = low 32 bits
// of the generic pointer (LDS aperture truncation, ISA §10.2).
// This toolchain exposes the 6-arg builtin (g0, g1, g2, g3, g4, cpol).
__device__ __forceinline__ void tdm_load_bf16_1d(const __bf16* gsrc,
                                                 const __bf16* lds_dst,
                                                 unsigned n_elem) {
  unsigned long long ga = (unsigned long long)(const void*)gsrc;
  unsigned lds_off = (unsigned)(unsigned long long)(const void*)lds_dst;
  u32x4 g0;
  g0[0] = 1u;                                            // count=1 (user D#)
  g0[1] = lds_off;                                       // lds_addr
  g0[2] = (unsigned)(ga & 0xFFFFFFFFu);                  // global_addr[31:0]
  g0[3] = (unsigned)((ga >> 32) & 0x1FFFFFFu) | (2u << 30); // [56:32] | type=2
  i32x8 g1;
  g1[0] = (int)(1u << 16);                               // data_size=1 (2 bytes)
  g1[1] = (int)((n_elem & 0xFFFFu) << 16);               // tensor_dim0[15:0]
  g1[2] = (int)((n_elem >> 16) & 0xFFFFu) | (int)(1u << 16); // dim0[31:16], dim1=1
  g1[3] = (int)((n_elem & 0xFFFFu) << 16);               // tile_dim0
  g1[4] = 1;                                             // tile_dim1=1
  g1[5] = (int)n_elem;                                   // dim0_stride[31:0]
  g1[6] = 0; g1[7] = 0;
  i32x4 z4 = {0, 0, 0, 0};
  i32x8 z8 = {0, 0, 0, 0, 0, 0, 0, 0};
  __builtin_amdgcn_tensor_load_to_lds(g0, g1, z4, z4, z8, 0);
}
#endif

__device__ __forceinline__ void wait_tensorcnt0() {
#ifdef HAVE_TENSORCNT
  __builtin_amdgcn_s_wait_tensorcnt(0);
#else
  asm volatile("s_wait_tensorcnt 0x0" ::: "memory");
#endif
}

// ---------------------------------------------------------------------------
// Weight packer: dst[Nout][Kpad] (bf16) = transpose of src[K][Nout] (f32),
// zero-padded K -> Kpad. Fragments become two contiguous 16B loads per lane.
// ---------------------------------------------------------------------------
__global__ void pack_wT_kernel(const float* __restrict__ src, __bf16* __restrict__ dst,
                               int K, int Nout, int Kpad) {
  int idx = blockIdx.x * blockDim.x + threadIdx.x;
  if (idx >= Nout * Kpad) return;
  int n = idx / Kpad, k = idx % Kpad;
  float v = (k < K) ? src[(size_t)k * Nout + n] : 0.f;
  dst[idx] = (__bf16)v;
}

__global__ void zero_f32_kernel(float* __restrict__ p, int n) {
  int i = blockIdx.x * blockDim.x + threadIdx.x;
  if (i < n) p[i] = 0.f;
}

// ---------------------------------------------------------------------------
// s = x @ emb_W + emb_b; also zero v / ds / dv for this node tile.
// ---------------------------------------------------------------------------
__global__ void embed_kernel(const float* __restrict__ x,
                             const __bf16* __restrict__ embWt,
                             const float* __restrict__ emb_b,
                             float* __restrict__ s, float* __restrict__ v,
                             float* __restrict__ ds, float* __restrict__ dv) {
  __shared__ __align__(16) __bf16 xl[16 * 96];
  const int tid = threadIdx.x;
  const int node0 = blockIdx.x * 16;
  for (int i = tid; i < 16 * 96; i += 256) {
    int m = i / 96, k = i % 96;
    float val = (k < 92) ? x[(size_t)(node0 + m) * 92 + k] : 0.f;
    xl[i] = (__bf16)val;
  }
  for (int i = tid; i < 16 * 384; i += 256) {
    v[(size_t)node0 * 384 + i]  = 0.f;
    dv[(size_t)node0 * 384 + i] = 0.f;
  }
  for (int i = tid; i < 16 * 128; i += 256) ds[(size_t)node0 * 128 + i] = 0.f;
  __syncthreads();

  const int w = tid >> 5, lane = tid & 31, half = lane >> 4, col = lane & 15;
  v8f acc = {};
#pragma unroll
  for (int kc = 0; kc < 96; kc += 32)
    acc = wmma_bf16(load_frag(xl, 96, 0, kc), load_frag(embWt, 96, w * 16, kc), acc);
  float bias = emb_b[w * 16 + col];
#pragma unroll
  for (int i = 0; i < 8; ++i) {
    int m = i + 8 * half;
    s[(size_t)(node0 + m) * 128 + w * 16 + col] = acc[i] + bias;
  }
}

// ---------------------------------------------------------------------------
// Fused message kernel: 32 edges / block, 8 waves, two M-tiles per wave so
// every B fragment feeds two WMMAs from registers.
//   h1  = silu(s[src] @ W1 + b1)   (W1 staged to LDS by TDM, overlapped with
//                                   the per-edge math + s[src] gather)
//   phi = h1 @ W2 + b2             (B streamed from L2)
//   Wch = (rbf @ Wr + br) * cutoff (K=32, rbf padded 20->32)
//   prod = phi * Wch; scatter dsm via f32 atomics;
//   dvm = v[src]*left + right*nrm scattered via f32 atomics.
// Wave w owns feature columns f = 16w+lane for all three 128-wide segments,
// so left/right never leave registers.
// ---------------------------------------------------------------------------
__global__ void message_kernel(const float* __restrict__ s, const float* __restrict__ v,
                               const int* __restrict__ ei, const float* __restrict__ ea,
                               const __bf16* __restrict__ l1Wt, const float* __restrict__ l1b,
                               const __bf16* __restrict__ l2Wt, const float* __restrict__ l2b,
                               const __bf16* __restrict__ rWt,  const float* __restrict__ rb,
                               float* __restrict__ ds, float* __restrict__ dv) {
  __shared__ __align__(16) __bf16 l1l[128 * 128];   // 32 KB, TDM-staged W1^T
  __shared__ __align__(16) __bf16 sl[32 * 128];     // 8 KB
  __shared__ __align__(16) __bf16 h1[32 * 128];     // 8 KB
  __shared__ __align__(16) __bf16 rbfl[32 * 32];    // 2 KB
  __shared__ int   srcl[32], dstl[32];
  __shared__ float cutl[32], nrml[32][3];

  const int tid = threadIdx.x;
  const int e0  = blockIdx.x * 32;
  const int w = tid >> 5, lane = tid & 31, half = lane >> 4, col = lane & 15;

#ifdef HAVE_TDM
  if (w == 0) tdm_load_bf16_1d(l1Wt, l1l, 128u * 128u);   // async DMA, wave 0
#else
  for (int i = tid * 8; i < 128 * 128; i += 256 * 8)
    *(bf16x8*)(l1l + i) = *(const bf16x8*)(l1Wt + i);
#endif

  if (tid < 32) {
    int e = e0 + tid;
    srcl[tid] = ei[e];
    dstl[tid] = ei[EE + e];
    float ax = ea[e * 3], ay = ea[e * 3 + 1], az = ea[e * 3 + 2];
    float d  = sqrtf(ax * ax + ay * ay + az * az);
    float sd = (d == 0.f) ? 1.f : d;
    const float w0 = 3.14159265358979323846f / 5.0f;
#pragma unroll
    for (int j = 0; j < 20; ++j)
      rbfl[tid * 32 + j] = (__bf16)(__sinf(d * (float)(j + 1) * w0) / sd);
#pragma unroll
    for (int j = 20; j < 32; ++j) rbfl[tid * 32 + j] = (__bf16)0.f;
    cutl[tid] = (d < 5.0f) ? 0.5f * (__cosf(d * w0) + 1.f) : 0.f;
    float invd = 1.f / fmaxf(d, 1e-12f);
    nrml[tid][0] = ax * invd; nrml[tid][1] = ay * invd; nrml[tid][2] = az * invd;
  }
  __syncthreads();   // srcl ready

  for (int i = tid; i < 32 * 128; i += 256) {
    int m = i >> 7, f = i & 127;
    sl[i] = (__bf16)s[(size_t)srcl[m] * 128 + f];
  }
#ifdef HAVE_TDM
  if (w == 0) wait_tensorcnt0();   // W1 tile landed in LDS
#endif
  __syncthreads();

  if (lane == 0) __builtin_prefetch(l2Wt + (size_t)w * 16 * 128, 0, 2);

  // h1 = silu(sl @ W1 + b1) : two M-tiles per wave, shared B fragment
  {
    v8f a0 = {}, a1 = {};
#pragma unroll
    for (int kc = 0; kc < 128; kc += 32) {
      v16bf b = load_frag(l1l, 128, w * 16, kc);
      a0 = wmma_bf16(load_frag(sl, 128, 0,  kc), b, a0);
      a1 = wmma_bf16(load_frag(sl, 128, 16, kc), b, a1);
    }
    float bias = l1b[w * 16 + col];
#pragma unroll
    for (int i = 0; i < 8; ++i) {
      int m = i + 8 * half;
      h1[m * 128        + w * 16 + col] = (__bf16)silu_f(a0[i] + bias);
      h1[(m + 16) * 128 + w * 16 + col] = (__bf16)silu_f(a1[i] + bias);
    }
  }
  __syncthreads();

  // three segment tiles: left (nt=w), dsm (nt=w+8), right (nt=w+16)
  v8f seg0[3], seg1[3];
#pragma unroll
  for (int t = 0; t < 3; ++t) {
    int n0 = (w + 8 * t) * 16;
    v8f a0 = {}, a1 = {};
#pragma unroll
    for (int kc = 0; kc < 128; kc += 32) {
      v16bf b = load_frag(l2Wt, 128, n0, kc);
      a0 = wmma_bf16(load_frag(h1, 128, 0,  kc), b, a0);
      a1 = wmma_bf16(load_frag(h1, 128, 16, kc), b, a1);
    }
    v8f r0 = {}, r1 = {};
    {
      v16bf rbm = load_frag(rWt, 32, n0, 0);
      r0 = wmma_bf16(load_frag(rbfl, 32, 0,  0), rbm, r0);
      r1 = wmma_bf16(load_frag(rbfl, 32, 16, 0), rbm, r1);
    }
    float bphi = l2b[n0 + col];
    float brbf = rb[n0 + col];
#pragma unroll
    for (int i = 0; i < 8; ++i) {
      int m = i + 8 * half;
      a0[i] = (a0[i] + bphi) * ((r0[i] + brbf) * cutl[m]);
      a1[i] = (a1[i] + bphi) * ((r1[i] + brbf) * cutl[m + 16]);
    }
    seg0[t] = a0; seg1[t] = a1;
  }

  const int f = w * 16 + col;   // same feature column for all three segments
#pragma unroll
  for (int mt = 0; mt < 2; ++mt) {
#pragma unroll
    for (int i = 0; i < 8; ++i) {
      int m = i + 8 * half + 16 * mt;
      float dsm = mt ? seg1[1][i] : seg0[1][i];
      float lft = mt ? seg1[0][i] : seg0[0][i];
      float rgt = mt ? seg1[2][i] : seg0[2][i];
      atomic_add_f32(&ds[(size_t)dstl[m] * 128 + f], dsm);
      size_t vsrc = (size_t)srcl[m] * 384 + (size_t)f * 3;
      size_t vdst = (size_t)dstl[m] * 384 + (size_t)f * 3;
#pragma unroll
      for (int k = 0; k < 3; ++k)
        atomic_add_f32(&dv[vdst + k], v[vsrc + k] * lft + rgt * nrml[m][k]);
    }
  }
}

// ---------------------------------------------------------------------------
// Fused update kernel: 16 nodes / block.
//   s1 = s + ds; v1 = v + dv
//   U=v1@UW, V=v1@VW (per k), UV=sum_k U*V, nV=|V|
//   a = silu([s1,nV]@upW+upb) @ l2W + b
//   s = s1 + mid*UV + bot ; v = v1*(1+top) ; zero ds/dv for next layer.
// ---------------------------------------------------------------------------
__global__ void update_kernel(float* __restrict__ s, float* __restrict__ v,
                              float* __restrict__ ds, float* __restrict__ dv,
                              const __bf16* __restrict__ UWt, const __bf16* __restrict__ VWt,
                              const __bf16* __restrict__ upWt, const float* __restrict__ upb,
                              const __bf16* __restrict__ l2Wt, const float* __restrict__ l2b) {
  __shared__ __align__(16) float  v1[16 * 384];
  __shared__ __align__(16) float  s1[16 * 128];
  __shared__ __align__(16) __bf16 vk[16 * 128];
  __shared__ __align__(16) __bf16 cat[16 * 256];
  __shared__ __align__(16) __bf16 hl[16 * 128];

  const int tid = threadIdx.x;
  const int node0 = blockIdx.x * 16;

  for (int i = tid; i < 16 * 128; i += 256) {
    float sv = s[(size_t)node0 * 128 + i] + ds[(size_t)node0 * 128 + i];
    s1[i] = sv;
    cat[(i >> 7) * 256 + (i & 127)] = (__bf16)sv;
  }
  for (int i = tid; i < 16 * 384; i += 256)
    v1[i] = v[(size_t)node0 * 384 + i] + dv[(size_t)node0 * 384 + i];
  __syncthreads();

  const int w = tid >> 5, lane = tid & 31, half = lane >> 4, col = lane & 15;

  v8f uv = {}, vv = {};
  for (int k = 0; k < 3; ++k) {
    for (int i = tid; i < 16 * 128; i += 256)
      vk[i] = (__bf16)v1[i * 3 + k];
    __syncthreads();
    v8f aU = {}, aV = {};
#pragma unroll
    for (int kc = 0; kc < 128; kc += 32) {
      v16bf a = load_frag(vk, 128, 0, kc);
      aU = wmma_bf16(a, load_frag(UWt, 128, w * 16, kc), aU);
      aV = wmma_bf16(a, load_frag(VWt, 128, w * 16, kc), aV);
    }
#pragma unroll
    for (int i = 0; i < 8; ++i) { uv[i] += aU[i] * aV[i]; vv[i] += aV[i] * aV[i]; }
    __syncthreads();
  }
#pragma unroll
  for (int i = 0; i < 8; ++i) {
    int m = i + 8 * half;
    cat[m * 256 + 128 + w * 16 + col] = (__bf16)sqrtf(vv[i]);
  }
  __syncthreads();

  // a1 = silu(cat @ upW + upb), K = 256
  {
    v8f acc = {};
#pragma unroll
    for (int kc = 0; kc < 256; kc += 32)
      acc = wmma_bf16(load_frag(cat, 256, 0, kc), load_frag(upWt, 256, w * 16, kc), acc);
    float bias = upb[w * 16 + col];
#pragma unroll
    for (int i = 0; i < 8; ++i) {
      int m = i + 8 * half;
      hl[m * 128 + w * 16 + col] = (__bf16)silu_f(acc[i] + bias);
    }
  }
  __syncthreads();

  // top (nt=w), mid (nt=w+8), bot (nt=w+16)
  v8f seg[3];
#pragma unroll
  for (int t = 0; t < 3; ++t) {
    int n0 = (w + 8 * t) * 16;
    v8f acc = {};
#pragma unroll
    for (int kc = 0; kc < 128; kc += 32)
      acc = wmma_bf16(load_frag(hl, 128, 0, kc), load_frag(l2Wt, 128, n0, kc), acc);
    float bias = l2b[n0 + col];
#pragma unroll
    for (int i = 0; i < 8; ++i) acc[i] += bias;
    seg[t] = acc;
  }

  const int f = w * 16 + col;   // uv/vv fragments align with seg fragments
#pragma unroll
  for (int i = 0; i < 8; ++i) {
    int m = i + 8 * half;
    size_t srow = (size_t)(node0 + m) * 128 + f;
    s[srow]  = s1[m * 128 + f] + seg[1][i] * uv[i] + seg[2][i];
    ds[srow] = 0.f;
    float scale = 1.f + seg[0][i];
    size_t vrow = (size_t)(node0 + m) * 384 + (size_t)f * 3;
#pragma unroll
    for (int k = 0; k < 3; ++k) {
      v[vrow + k]  = v1[(m * 128 + f) * 3 + k] * scale;
      dv[vrow + k] = 0.f;
    }
  }
}

// ---------------------------------------------------------------------------
// Output head: o = silu(s@W1+b1)@W2+b2 ; atomics into per-graph sum.
// ---------------------------------------------------------------------------
__global__ void output_kernel(const float* __restrict__ s, const int* __restrict__ batch,
                              const __bf16* __restrict__ o1Wt, const float* __restrict__ o1b,
                              const __bf16* __restrict__ o2Wt, const float* __restrict__ o2b,
                              float* __restrict__ out) {
  __shared__ __align__(16) __bf16 sl[16 * 128];
  __shared__ __align__(16) __bf16 hl[16 * 128];
  __shared__ int bl[16];
  const int tid = threadIdx.x;
  const int node0 = blockIdx.x * 16;
  if (tid < 16) bl[tid] = batch[node0 + tid];
  for (int i = tid; i < 16 * 128; i += 256) sl[i] = (__bf16)s[(size_t)node0 * 128 + i];
  __syncthreads();

  const int w = tid >> 5, lane = tid & 31, half = lane >> 4, col = lane & 15;
  {
    v8f acc = {};
#pragma unroll
    for (int kc = 0; kc < 128; kc += 32)
      acc = wmma_bf16(load_frag(sl, 128, 0, kc), load_frag(o1Wt, 128, w * 16, kc), acc);
    float bias = o1b[w * 16 + col];
#pragma unroll
    for (int i = 0; i < 8; ++i) {
      int m = i + 8 * half;
      hl[m * 128 + w * 16 + col] = (__bf16)silu_f(acc[i] + bias);
    }
  }
  __syncthreads();
  {
    v8f acc = {};
#pragma unroll
    for (int kc = 0; kc < 128; kc += 32)
      acc = wmma_bf16(load_frag(hl, 128, 0, kc), load_frag(o2Wt, 128, w * 16, kc), acc);
    float bias = o2b[w * 16 + col];
#pragma unroll
    for (int i = 0; i < 8; ++i) {
      int m = i + 8 * half;
      atomic_add_f32(&out[(size_t)bl[m] * 128 + w * 16 + col], acc[i] + bias);
    }
  }
}

// ---------------------------------------------------------------------------
// Host launch
// ---------------------------------------------------------------------------
static inline size_t align256(size_t x) { return (x + 255) & ~(size_t)255; }

extern "C" void kernel_launch(void* const* d_in, const int* in_sizes, int n_in,
                              void* d_out, int out_size, void* d_ws, size_t ws_size,
                              hipStream_t stream) {
  (void)in_sizes; (void)n_in; (void)out_size; (void)ws_size;
  const float* x        = (const float*)d_in[0];
  const int*   ei       = (const int*)  d_in[1];
  const float* ea       = (const float*)d_in[2];
  const int*   batch    = (const int*)  d_in[3];
  const float* emb_W    = (const float*)d_in[4];
  const float* emb_b    = (const float*)d_in[5];
  const float* m1W      = (const float*)d_in[6];
  const float* m1b      = (const float*)d_in[7];
  const float* m2W      = (const float*)d_in[8];
  const float* m2b      = (const float*)d_in[9];
  const float* mrW      = (const float*)d_in[10];
  const float* mrb      = (const float*)d_in[11];
  const float* upW      = (const float*)d_in[12];
  const float* upb      = (const float*)d_in[13];
  const float* uUW      = (const float*)d_in[14];
  const float* uVW      = (const float*)d_in[15];
  const float* ul2W     = (const float*)d_in[16];
  const float* ul2b     = (const float*)d_in[17];
  const float* o1W      = (const float*)d_in[18];
  const float* o1b      = (const float*)d_in[19];
  const float* o2W      = (const float*)d_in[20];
  const float* o2b      = (const float*)d_in[21];

  char* ws = (char*)d_ws;
  size_t off = 0;
  float* s  = (float*)(ws + off); off = align256(off + (size_t)NN * FF * 4);
  float* v  = (float*)(ws + off); off = align256(off + (size_t)NN * FF * 3 * 4);
  float* ds = (float*)(ws + off); off = align256(off + (size_t)NN * FF * 4);
  float* dv = (float*)(ws + off); off = align256(off + (size_t)NN * FF * 3 * 4);

  auto alloc_bf = [&](size_t elems) {
    __bf16* p = (__bf16*)(ws + off);
    off = align256(off + elems * sizeof(__bf16));
    return p;
  };
  __bf16* embWt = alloc_bf(128 * 96);
  __bf16* o1Wt  = alloc_bf(128 * 128);
  __bf16* o2Wt  = alloc_bf(128 * 128);
  __bf16 *l1Wt[3], *l2Wt[3], *rWt[3], *upWt[3], *UWt[3], *VWt[3], *ul2Wt[3];
  for (int i = 0; i < 3; ++i) {
    l1Wt[i]  = alloc_bf(128 * 128);
    l2Wt[i]  = alloc_bf(384 * 128);
    rWt[i]   = alloc_bf(384 * 32);
    upWt[i]  = alloc_bf(128 * 256);
    UWt[i]   = alloc_bf(128 * 128);
    VWt[i]   = alloc_bf(128 * 128);
    ul2Wt[i] = alloc_bf(384 * 128);
  }

  auto pack = [&](const float* src, __bf16* dst, int K, int Nout, int Kpad) {
    int total = Nout * Kpad;
    pack_wT_kernel<<<(total + 255) / 256, 256, 0, stream>>>(src, dst, K, Nout, Kpad);
  };
  pack(emb_W, embWt, 92, 128, 96);
  pack(o1W, o1Wt, 128, 128, 128);
  pack(o2W, o2Wt, 128, 128, 128);
  for (int i = 0; i < 3; ++i) {
    pack(m1W  + (size_t)i * 128 * 128, l1Wt[i],  128, 128, 128);
    pack(m2W  + (size_t)i * 128 * 384, l2Wt[i],  128, 384, 128);
    pack(mrW  + (size_t)i * 20  * 384, rWt[i],   20,  384, 32);
    pack(upW  + (size_t)i * 256 * 128, upWt[i],  256, 128, 256);
    pack(uUW  + (size_t)i * 128 * 128, UWt[i],   128, 128, 128);
    pack(uVW  + (size_t)i * 128 * 128, VWt[i],   128, 128, 128);
    pack(ul2W + (size_t)i * 128 * 384, ul2Wt[i], 128, 384, 128);
  }

  zero_f32_kernel<<<(GG * FF + 255) / 256, 256, 0, stream>>>((float*)d_out, GG * FF);

  embed_kernel<<<NN / 16, 256, 0, stream>>>(x, embWt, emb_b, s, v, ds, dv);

  for (int i = 0; i < 3; ++i) {
    message_kernel<<<EE / 32, 256, 0, stream>>>(
        s, v, ei, ea,
        l1Wt[i], m1b + (size_t)i * 128,
        l2Wt[i], m2b + (size_t)i * 384,
        rWt[i],  mrb + (size_t)i * 384,
        ds, dv);
    update_kernel<<<NN / 16, 256, 0, stream>>>(
        s, v, ds, dv,
        UWt[i], VWt[i],
        upWt[i], upb + (size_t)i * 128,
        ul2Wt[i], ul2b + (size_t)i * 384);
  }

  output_kernel<<<NN / 16, 256, 0, stream>>>(s, batch, o1Wt, o1b, o2Wt, o2b,
                                             (float*)d_out);
}
